// GammaConnectivityPredictor_38259568672831
// MI455X (gfx1250) — compile-verified, matrix operand order checked by reading, not run on hardware
//
#include <hip/hip_runtime.h>
#include <hip/hip_bf16.h>
#include <stdint.h>

typedef __attribute__((ext_vector_type(16))) _Float16 v16h;
typedef __attribute__((ext_vector_type(8)))  _Float16 v8h;
typedef __attribute__((ext_vector_type(4)))  _Float16 v4h;
typedef __attribute__((ext_vector_type(8)))  float    v8f;

#define E_   512
#define C_   19
#define H_   8
#define HD_  64
#define NB_  16      // batch
#define NT_  512     // time steps
#define CP_  32      // tokens padded to 2 WMMA M-tiles
#define SX_  528     // LDS row stride in halves (1056 B -> every row 16B aligned)
#define SSROW_ 20    // score row stride in floats (valid 19x19 region only)

#define SHUF16(lo, hi) __builtin_shufflevector((lo), (hi), 0,1,2,3,4,5,6,7,8,9,10,11,12,13,14,15)

// ---------------------------------------------------------------------------
// Kernel 1: convert weights f32 -> f16 into workspace; zero batch accumulators
// ---------------------------------------------------------------------------
__global__ void gcp_prep_kernel(const float* __restrict__ wq,
                                const float* __restrict__ wk,
                                _Float16* __restrict__ wq16,
                                _Float16* __restrict__ wk16,
                                float* __restrict__ acc) {
    int i = blockIdx.x * blockDim.x + threadIdx.x;
    const int n = E_ * E_;
    for (int idx = i; idx < n; idx += gridDim.x * blockDim.x) {
        wq16[idx] = (_Float16)wq[idx];
        wk16[idx] = (_Float16)wk[idx];
    }
    if (blockIdx.x == 0 && threadIdx.x < NB_) acc[threadIdx.x] = 0.0f;
}

// ---------------------------------------------------------------------------
// Kernel 2: one block per (b,t). Async-staged X, fused QK projection (WMMA,
// both M-tiles share one B fragment), per-head scores (WMMA), softmax +
// strict-upper-triangle mean, atomically accumulated per batch.
// ---------------------------------------------------------------------------
__launch_bounds__(256)
__global__ void gcp_attn_kernel(const float* __restrict__ x,
                                const _Float16* __restrict__ wq16,
                                const _Float16* __restrict__ wk16,
                                const float* __restrict__ bq,
                                const float* __restrict__ bk,
                                float* __restrict__ acc) {
    __shared__ float    Xf[C_ * E_];              // 38,912 B (raw f32 stage)
    __shared__ _Float16 Xs[CP_ * SX_];            // 33,792 B
    __shared__ _Float16 Qs[CP_ * SX_];            // 33,792 B
    __shared__ _Float16 Ks[CP_ * SX_];            // 33,792 B
    __shared__ float    Sb[H_ * C_ * SSROW_];     // 12,160 B
    __shared__ float    red[256];                 //  1,024 B

    const int tid  = threadIdx.x;
    const int lane = tid & 31;
    const int wave = tid >> 5;
    const int l    = lane & 15;
    const int hl   = lane >> 4;   // half-wave selector (K-split for A/B frags)

    // -------- Phase A: async-copy X[19,512] f32 -> LDS (ASYNCcnt path),
    //          then convert to f16 padded operand buffer Xs.
    {
        const float* gbase = x + (size_t)blockIdx.x * (C_ * E_);
        for (int idx = tid; idx < (C_ * E_) / 4; idx += 256) {
            uint32_t ldsa = (uint32_t)(uintptr_t)(&Xf[idx * 4]);
            uint64_t ga   = (uint64_t)(uintptr_t)(gbase + idx * 4);
            asm volatile("global_load_async_to_lds_b128 %0, %1, off"
                         :: "v"(ldsa), "v"(ga) : "memory");
        }
        asm volatile("s_wait_asynccnt 0" ::: "memory");
        __syncthreads();

        for (int idx = tid; idx < C_ * (E_ / 4); idx += 256) {
            int r  = idx >> 7;        // / (512/4)
            int c4 = idx & 127;
            float4 v = *reinterpret_cast<const float4*>(&Xf[idx * 4]);
            v4h h;
            h[0] = (_Float16)v.x; h[1] = (_Float16)v.y;
            h[2] = (_Float16)v.z; h[3] = (_Float16)v.w;
            *reinterpret_cast<v4h*>(&Xs[r * SX_ + (c4 << 2)]) = h;
        }
        for (int idx = tid; idx < (CP_ - C_) * SX_; idx += 256)
            Xs[C_ * SX_ + idx] = (_Float16)0.0f;
    }
    __syncthreads();

    // -------- Phase B: Q = X*Wq^T + bq, K = X*Wk^T + bk (f16 WMMA, f32 acc)
    // 64 (proj, N-tile) pairs; each wave accumulates BOTH M-tiles against a
    // single shared B fragment (halves weight traffic vs. per-(mt) tiles).
    for (int i = 0; i < 8; ++i) {
        const int pt = wave + (i << 3);      // 0..63
        const int p  = pt >> 5;              // 0 = Q, 1 = K
        const int nt = pt & 31;
        const _Float16* W    = p ? wk16 : wq16;
        const float*    bias = p ? bk : bq;
        _Float16*       Out  = p ? Ks : Qs;
        const int nbase = nt << 4;

        v8f c0 = {};                          // rows 0..15
        v8f c1 = {};                          // rows 16..31
        const int arow0 = l * SX_;
        const int arow1 = (16 + l) * SX_;
        const _Float16* wrow = W + (size_t)(nbase + l) * E_;  // B = W^T rows = N
        for (int kb = 0; kb < 16; ++kb) {
            const int k0 = (kb << 5) + (hl << 3);
            __builtin_prefetch(wrow + k0 + 64, 0, 3);  // global_prefetch_b8
            v8h b0 = *reinterpret_cast<const v8h*>(wrow + k0);
            v8h b1 = *reinterpret_cast<const v8h*>(wrow + k0 + 16);
            v16h Bf = SHUF16(b0, b1);
            v8h a00 = *reinterpret_cast<const v8h*>(&Xs[arow0 + k0]);
            v8h a01 = *reinterpret_cast<const v8h*>(&Xs[arow0 + k0 + 16]);
            v8h a10 = *reinterpret_cast<const v8h*>(&Xs[arow1 + k0]);
            v8h a11 = *reinterpret_cast<const v8h*>(&Xs[arow1 + k0 + 16]);
            v16h A0 = SHUF16(a00, a01);
            v16h A1 = SHUF16(a10, a11);
            c0 = __builtin_amdgcn_wmma_f32_16x16x32_f16(
                     false, A0, false, Bf, (short)0, c0, false, false);
            c1 = __builtin_amdgcn_wmma_f32_16x16x32_f16(
                     false, A1, false, Bf, (short)0, c1, false, false);
        }
        const int   col = nbase + l;
        const float bv  = bias[col];
        #pragma unroll
        for (int r = 0; r < 8; ++r) {
            const int m = (hl << 3) + r;      // C/D layout: lanes16-31 = M 8..15
            Out[m * SX_ + col]        = (_Float16)(c0[r] + bv);
            Out[(16 + m) * SX_ + col] = (_Float16)(c1[r] + bv);
        }
    }
    __syncthreads();

    // -------- Phase C: per-head scores S_h = Q_h * K_h^T  (one wave per head)
    {
        const int h  = wave;
        const int hb = h * HD_;
        #pragma unroll
        for (int t2 = 0; t2 < 4; ++t2) {
            const int mt = t2 >> 1, nt = t2 & 1;
            v8f c = {};
            const int qrow = ((mt << 4) + l) * SX_ + hb;
            const int krow = ((nt << 4) + l) * SX_ + hb;  // B = K^T -> rows = N
            #pragma unroll
            for (int kb = 0; kb < 2; ++kb) {
                const int k0 = (kb << 5) + (hl << 3);
                v8h a0 = *reinterpret_cast<const v8h*>(&Qs[qrow + k0]);
                v8h a1 = *reinterpret_cast<const v8h*>(&Qs[qrow + k0 + 16]);
                v8h b0 = *reinterpret_cast<const v8h*>(&Ks[krow + k0]);
                v8h b1 = *reinterpret_cast<const v8h*>(&Ks[krow + k0 + 16]);
                v16h A  = SHUF16(a0, a1);
                v16h Bm = SHUF16(b0, b1);
                c = __builtin_amdgcn_wmma_f32_16x16x32_f16(
                        false, A, false, Bm, (short)0, c, false, false);
            }
            const int n = (nt << 4) + l;
            if (n < C_) {
                #pragma unroll
                for (int r = 0; r < 8; ++r) {
                    const int m = (mt << 4) + (hl << 3) + r;
                    if (m < C_)
                        Sb[h * (C_ * SSROW_) + m * SSROW_ + n] = c[r];
                }
            }
        }
    }
    __syncthreads();

    // -------- Phase D: row softmax (over k), strict-upper partial sums
    float partial = 0.0f;
    if (tid < H_ * C_) {
        const int h = tid / C_;
        const int q = tid % C_;
        const float* srow = &Sb[h * (C_ * SSROW_) + q * SSROW_];
        const float scale = 0.125f;           // 1/sqrt(HD)
        float mx = -1e30f;
        for (int k = 0; k < C_; ++k) mx = fmaxf(mx, srow[k] * scale);
        float denom = 0.0f, up = 0.0f;
        for (int k = 0; k < C_; ++k) {
            float e = __expf(srow[k] * scale - mx);
            denom += e;
            if (k > q) up += e;
        }
        partial = up / denom;
    }
    red[tid] = partial;
    __syncthreads();
    for (int s = 128; s > 0; s >>= 1) {
        if (tid < s) red[tid] += red[tid + s];
        __syncthreads();
    }
    if (tid == 0) {
        // strength(b,t) = sum_upper(mean_h attn) / 171 = red0 / (H * 171)
        const float strength = red[0] / (float)(H_ * (C_ * (C_ - 1) / 2));
        const int b = blockIdx.x / NT_;
        atomicAdd(&acc[b], strength * (1.0f / (float)NT_));
    }
}

// ---------------------------------------------------------------------------
// Kernel 3: clamp(relu(mean_t)) -> out[B,1]
// ---------------------------------------------------------------------------
__global__ void gcp_finalize_kernel(const float* __restrict__ acc,
                                    float* __restrict__ out) {
    int b = threadIdx.x;
    if (b < NB_) {
        float v = acc[b];
        v = v < 0.0f ? 0.0f : v;
        v = v > 1.0f ? 1.0f : v;
        out[b] = v;
    }
}

extern "C" void kernel_launch(void* const* d_in, const int* in_sizes, int n_in,
                              void* d_out, int out_size, void* d_ws, size_t ws_size,
                              hipStream_t stream) {
    const float* x  = (const float*)d_in[0];   // [B,T,C,E]
    const float* wq = (const float*)d_in[1];   // [E,E]
    const float* wk = (const float*)d_in[2];   // [E,E]
    const float* bq = (const float*)d_in[3];   // [E]
    const float* bk = (const float*)d_in[4];   // [E]
    float* out = (float*)d_out;                // [B,1]

    _Float16* wq16 = (_Float16*)d_ws;
    _Float16* wk16 = wq16 + E_ * E_;
    float*    acc  = (float*)(wk16 + E_ * E_); // 16 floats

    gcp_prep_kernel<<<256, 256, 0, stream>>>(wq, wk, wq16, wk16, acc);
    gcp_attn_kernel<<<NB_ * NT_, 256, 0, stream>>>(x, wq16, wk16, bq, bk, acc);
    gcp_finalize_kernel<<<1, 32, 0, stream>>>(acc, out);
}
